// STTransformerCatNoCls_50972671869554
// MI455X (gfx1250) — matrix operand order repaired and checked by reading, not compiled, not run on hardware
//
#include <hip/hip_runtime.h>
#include <hip/hip_bf16.h>

typedef unsigned short u16;
typedef unsigned long long ull;
typedef __attribute__((ext_vector_type(16))) __bf16          v16bf;
typedef __attribute__((ext_vector_type(8)))  float           v8f;
typedef __attribute__((ext_vector_type(8)))  unsigned short  us8;

union ABfrag { us8 h[2]; v16bf v; };

__device__ __forceinline__ u16 f2bf(float f) {
  unsigned u = __float_as_uint(f);
  u = (u + 0x7FFFu + ((u >> 16) & 1u)) >> 16;
  return (u16)u;
}
__device__ __forceinline__ float bf2f(u16 h) {
  return __uint_as_float(((unsigned)h) << 16);
}
__device__ __forceinline__ float gelu_exact(float a) {
  return 0.5f * a * (1.0f + erff(a * 0.70710678118654752f));
}

// ---------------------------------------------------------------------------
// Elementwise helpers
// ---------------------------------------------------------------------------
__global__ void wcast_kernel(const float* __restrict__ in, u16* __restrict__ out, int n) {
  int i = blockIdx.x * blockDim.x + threadIdx.x;
  if (i < n) out[i] = f2bf(in[i]);
}

__global__ void copyf_kernel(const float* __restrict__ in, float* __restrict__ out, int n) {
  int i = blockIdx.x * blockDim.x + threadIdx.x;
  if (i < n) out[i] = in[i];
}

__global__ void zerof_kernel(float* __restrict__ p, int n) {
  int i = blockIdx.x * blockDim.x + threadIdx.x;
  if (i < n) p[i] = 0.f;
}

__global__ void castscale_kernel(const float* __restrict__ in, u16* __restrict__ out,
                                 float s, int n) {
  int i = blockIdx.x * blockDim.x + threadIdx.x;
  if (i < n) out[i] = f2bf(in[i] * s);
}

// ---------------------------------------------------------------------------
// LayerNorm over 256 cols, one wave per row, writes bf16
// ---------------------------------------------------------------------------
__global__ void ln_cast_kernel(const float* __restrict__ x, const float* __restrict__ g,
                               const float* __restrict__ b, u16* __restrict__ out, int rows) {
  int wave = (blockIdx.x * blockDim.x + threadIdx.x) >> 5;
  int lane = threadIdx.x & 31;
  if (wave >= rows) return;
  const float* xr = x + (ull)wave * 256;
  float v[8]; float s = 0.f;
  #pragma unroll
  for (int i = 0; i < 8; ++i) { v[i] = xr[lane + 32 * i]; s += v[i]; }
  #pragma unroll
  for (int o = 16; o > 0; o >>= 1) s += __shfl_xor(s, o, 32);
  float mean = s * (1.f / 256.f);
  float s2 = 0.f;
  #pragma unroll
  for (int i = 0; i < 8; ++i) { float d = v[i] - mean; s2 += d * d; }
  #pragma unroll
  for (int o = 16; o > 0; o >>= 1) s2 += __shfl_xor(s2, o, 32);
  float rstd = rsqrtf(s2 * (1.f / 256.f) + 1e-5f);
  u16* orow = out + (ull)wave * 256;
  #pragma unroll
  for (int i = 0; i < 8; ++i) {
    int c = lane + 32 * i;
    orow[c] = f2bf((v[i] - mean) * rstd * g[c] + b[c]);
  }
}

// ---------------------------------------------------------------------------
// Generic batched GEMM:  C[m,n] = scale * sum_k A[m,k]*W[n,k]  (+bias)(+resid)
// A bf16 [M,K] rowmajor (lda), W bf16 [N(+N gate rows if GEGLU),K] rowmajor (ldw).
// Wave computes a (16*RT)x64 tile. RT<=2 keeps accumulators+frags in registers
// (RT=4 was observed to spill C operands to scratch on gfx1250).
// GEGLU (RT must be 1): gate columns are W rows n+1024, bias[n+1024];
//                       writes gelu(a)*g bf16.
// ATOMIC: split-K partials accumulated into f32 C via atomicAdd (gridDim.z
//         selects the K-chunk; K is the chunk length).
// ---------------------------------------------------------------------------
template <int RT, bool GEGLU, bool ATOMIC>
__global__ void gemm_bf16_wmma(
    const u16* __restrict__ A, const u16* __restrict__ W,
    const float* __restrict__ bias, const float* __restrict__ resid,
    float* __restrict__ Cf, u16* __restrict__ Cb,
    int M, int N, int K, int lda, int ldw, int ldc,
    ull sA, ull sW, int cInner, ull sCo, ull sCi, float scale) {
  const int lane = threadIdx.x & 31;
  const int wid  = (blockIdx.x * blockDim.x + threadIdx.x) >> 5;
  const int tilesN = N >> 6;
  const int tiles  = (M / (16 * RT)) * tilesN;
  if (wid >= tiles) return;
  const int batch = blockIdx.y;
  const int tm = (wid / tilesN) * (16 * RT);
  const int tn = (wid % tilesN) << 6;
  const int half = lane >> 4;
  const int r = lane & 15;
  const ull koff = (ull)blockIdx.z * (ull)K;   // split-K chunk offset

  const u16* Ab = A + (ull)batch * sA;
  const u16* Wb = W + (ull)batch * sW;
  const ull cOff = (ull)(batch / cInner) * sCo + (ull)(batch % cInner) * sCi;

  const u16* arow = Ab + (ull)(tm + r) * lda + half * 8 + koff;
  const u16* wrow[4];
  const u16* grow[4];
  #pragma unroll
  for (int j = 0; j < 4; ++j) {
    wrow[j] = Wb + (ull)(tn + 16 * j + r) * ldw + half * 8 + koff;
    if (GEGLU) grow[j] = Wb + (ull)(1024 + tn + 16 * j + r) * ldw + half * 8 + koff;
  }

  v8f acc[RT][4];
  v8f accg[GEGLU ? RT : 1][4];
  #pragma unroll
  for (int t = 0; t < RT; ++t)
    #pragma unroll
    for (int j = 0; j < 4; ++j) {
      acc[t][j] = (v8f){};
      if (GEGLU) accg[t][j] = (v8f){};
    }

  for (int k0 = 0; k0 < K; k0 += 32) {
    ABfrag a[RT];
    #pragma unroll
    for (int t = 0; t < RT; ++t) {
      a[t].h[0] = *(const us8*)(arow + (ull)t * 16 * lda + k0);
      a[t].h[1] = *(const us8*)(arow + (ull)t * 16 * lda + k0 + 16);
    }
    #pragma unroll
    for (int j = 0; j < 4; ++j) {
      ABfrag bf; bf.h[0] = *(const us8*)(wrow[j] + k0); bf.h[1] = *(const us8*)(wrow[j] + k0 + 16);
      v16bf bj = bf.v;
      #pragma unroll
      for (int t = 0; t < RT; ++t)
        acc[t][j] = __builtin_amdgcn_wmma_f32_16x16x32_bf16(
            false, a[t].v, false, bj, (short)0, acc[t][j], false, false);
      if (GEGLU) {
        ABfrag gf; gf.h[0] = *(const us8*)(grow[j] + k0); gf.h[1] = *(const us8*)(grow[j] + k0 + 16);
        v16bf gj = gf.v;
        #pragma unroll
        for (int t = 0; t < RT; ++t)
          accg[t][j] = __builtin_amdgcn_wmma_f32_16x16x32_bf16(
              false, a[t].v, false, gj, (short)0, accg[t][j], false, false);
      }
    }
  }

  #pragma unroll
  for (int t = 0; t < RT; ++t) {
    #pragma unroll
    for (int j = 0; j < 4; ++j) {
      const int n = tn + 16 * j + r;
      const float bv = (!ATOMIC && bias) ? bias[n] : 0.f;
      const float bg = (GEGLU && bias) ? bias[1024 + n] : 0.f;
      #pragma unroll
      for (int i = 0; i < 8; ++i) {
        const int m = tm + 16 * t + i + 8 * half;
        const ull idx = cOff + (ull)m * ldc + n;
        if (ATOMIC) {
          atomicAdd(&Cf[idx], acc[t][j][i]);
        } else if (GEGLU) {
          float av = acc[t][j][i] * scale + bv;
          float gv = accg[t][j][i] * scale + bg;
          Cb[idx] = f2bf(gelu_exact(av) * gv);
        } else {
          float val = acc[t][j][i] * scale + bv;
          if (resid) val += resid[idx];
          if (Cf) Cf[idx] = val; else Cb[idx] = f2bf(val);
        }
      }
    }
  }
}

// ---------------------------------------------------------------------------
// Instance-norm on K,V (over 64) + 2D RoPE on Q,K. One wave per (b,n,h).
// qkv bf16 [B*N,1536] -> q [B*H,N,64], kT/vT transposed [B*H,64,N] (bf16)
// ---------------------------------------------------------------------------
__global__ void instnorm_rope_kernel(const u16* __restrict__ qkv, const float* __restrict__ pos,
                                     u16* __restrict__ qo, u16* __restrict__ kT,
                                     u16* __restrict__ vT) {
  const int lane = threadIdx.x & 31;
  const int gw = (blockIdx.x * blockDim.x + threadIdx.x) >> 5;  // ((b*4096+n)*8+h)
  const int h = gw & 7;
  const int bn = gw >> 3;
  const int n = bn & 4095;
  const int b = bn >> 12;
  const u16* row = qkv + (ull)bn * 1536;
  float q0 = bf2f(row[h * 64 + lane]);
  float q1 = bf2f(row[h * 64 + lane + 32]);
  float k0 = bf2f(row[512 + h * 64 + lane]);
  float k1 = bf2f(row[512 + h * 64 + lane + 32]);
  float v0 = bf2f(row[1024 + h * 64 + lane]);
  float v1 = bf2f(row[1024 + h * 64 + lane + 32]);

  float ks = k0 + k1, ks2 = k0 * k0 + k1 * k1;
  float vs = v0 + v1, vs2 = v0 * v0 + v1 * v1;
  #pragma unroll
  for (int o = 16; o > 0; o >>= 1) {
    ks  += __shfl_xor(ks,  o, 32);
    ks2 += __shfl_xor(ks2, o, 32);
    vs  += __shfl_xor(vs,  o, 32);
    vs2 += __shfl_xor(vs2, o, 32);
  }
  float km = ks * (1.f / 64.f);
  float kr = rsqrtf(fmaxf(ks2 * (1.f / 64.f) - km * km, 0.f) + 1e-5f);
  float vm = vs * (1.f / 64.f);
  float vr = rsqrtf(fmaxf(vs2 * (1.f / 64.f) - vm * vm, 0.f) + 1e-5f);
  k0 = (k0 - km) * kr; k1 = (k1 - km) * kr;
  v0 = (v0 - vm) * vr; v1 = (v1 - vm) * vr;

  // 2D rotary: d<32 uses x coord, d>=32 uses y; within 32, freqs concat [f,f]
  const float cx = pos[(ull)bn * 2 + 0];
  const float cy = pos[(ull)bn * 2 + 1];
  const float invf = __expf(-(float)(lane & 15) * 0.57564627324851147f); // ln(1e4)/16
  const float fx = cx * 2048.f * invf;   // ROPE_FACTOR = 32*64
  const float fy = cy * 2048.f * invf;
  const float sgn = (lane < 16) ? -1.f : 1.f;
  float pq0 = __shfl_xor(q0, 16, 32), pq1 = __shfl_xor(q1, 16, 32);
  float pk0 = __shfl_xor(k0, 16, 32), pk1 = __shfl_xor(k1, 16, 32);
  float cosx = cosf(fx), sinx = sinf(fx), cosy = cosf(fy), siny = sinf(fy);
  float qr0 = q0 * cosx + sgn * pq0 * sinx;
  float qr1 = q1 * cosy + sgn * pq1 * siny;
  float kr0 = k0 * cosx + sgn * pk0 * sinx;
  float kr1 = k1 * cosy + sgn * pk1 * siny;

  const ull bh = (ull)(b * 8 + h);
  u16* qrow = qo + (bh * 4096 + n) * 64;
  qrow[lane] = f2bf(qr0);
  qrow[lane + 32] = f2bf(qr1);
  u16* kc = kT + bh * 262144;  // 64*4096
  kc[(ull)lane * 4096 + n] = f2bf(kr0);
  kc[(ull)(lane + 32) * 4096 + n] = f2bf(kr1);
  u16* vc = vT + bh * 262144;
  vc[(ull)lane * 4096 + n] = f2bf(v0);
  vc[(ull)(lane + 32) * 4096 + n] = f2bf(v1);
}

// ---------------------------------------------------------------------------
extern "C" void kernel_launch(void* const* d_in, const int* in_sizes, int n_in,
                              void* d_out, int out_size, void* d_ws, size_t ws_size,
                              hipStream_t stream) {
  (void)in_sizes; (void)n_in; (void)out_size; (void)ws_size;
  const float* x_in  = (const float*)d_in[0];
  const float* pos   = (const float*)d_in[1];
  const float* Wqkv  = (const float*)d_in[2];
  const float* Wout  = (const float*)d_in[3];
  const float* bout  = (const float*)d_in[4];
  const float* ln1g  = (const float*)d_in[5];
  const float* ln1b  = (const float*)d_in[6];
  const float* ln2g  = (const float*)d_in[7];
  const float* ln2b  = (const float*)d_in[8];
  const float* W1    = (const float*)d_in[9];
  const float* b1    = (const float*)d_in[10];
  const float* W2    = (const float*)d_in[11];
  const float* b2    = (const float*)d_in[12];
  float* xbuf = (float*)d_out;  // residual stream lives in d_out

  const int B = 8, H = 8, R = 8 * 4096;  // R = 32768
  char* ws = (char*)d_ws;
  // workspace layout (bytes)
  u16*   wq_bf   = (u16*)(ws + 0);                //  4*1536*256 bf16
  u16*   wo_bf   = (u16*)(ws + 3145728ull);       //  4*256*512
  u16*   w1_bf   = (u16*)(ws + 4194304ull);       //  4*2048*256
  u16*   w2_bf   = (u16*)(ws + 8388608ull);       //  4*256*1024
  u16*   ln_bf   = (u16*)(ws + 10485760ull);      //  R*256
  u16*   big_bf  = (u16*)(ws + 27262976ull);      //  R*1536 (qkv)
  u16*   q_bf    = (u16*)(ws + 161480704ull);     //  R*512
  u16*   kT_bf   = (u16*)(ws + 195035136ull);     //  R*512
  u16*   vT_bf   = (u16*)(ws + 228589568ull);     //  R*512
  u16*   attn_bf = (u16*)(ws + 262144000ull);     //  R*512
  u16*   dots_bf = (u16*)(ws + 295698432ull);     //  B*H*64*64 bf16
  u16*   gg_bf   = (u16*)(ws + 296222720ull);     //  R*1024
  float* dots_f  = (float*)(ws + 363331584ull);   //  B*H*64*64 f32 (split-K acc)

  // weights -> bf16 (once per launch; cheap)
  wcast_kernel<<<(1572864 + 255) / 256, 256, 0, stream>>>(Wqkv, wq_bf, 1572864);
  wcast_kernel<<<(524288  + 255) / 256, 256, 0, stream>>>(Wout, wo_bf, 524288);
  wcast_kernel<<<(2097152 + 255) / 256, 256, 0, stream>>>(W1,   w1_bf, 2097152);
  wcast_kernel<<<(1048576 + 255) / 256, 256, 0, stream>>>(W2,   w2_bf, 1048576);
  // residual stream init
  copyf_kernel<<<(R * 256 + 255) / 256, 256, 0, stream>>>(x_in, xbuf, R * 256);

  for (int l = 0; l < 4; ++l) {
    // ---- attention block ----
    ln_cast_kernel<<<R * 32 / 256, 256, 0, stream>>>(xbuf, ln1g + l * 256, ln1b + l * 256,
                                                     ln_bf, R);
    // qkv = ln @ Wqkv^T : [R,256]x[1536,256] -> bf16 [R,1536]   (32x64 wave tiles)
    gemm_bf16_wmma<2, false, false><<<dim3((R / 32) * (1536 / 64) / 8, 1), 256, 0, stream>>>(
        ln_bf, wq_bf + (ull)l * 393216, nullptr, nullptr, nullptr, big_bf,
        R, 1536, 256, 256, 256, 1536, 0ull, 0ull, 1, 0ull, 0ull, 1.f);
    instnorm_rope_kernel<<<R * H * 32 / 256, 256, 0, stream>>>(big_bf, pos, q_bf, kT_bf, vT_bf);
    // dotsT[e,d] = sum_n v[n,e]*k[n,d] : A=vT, W=kT, per (b,h); split-K=8, f32 atomics
    zerof_kernel<<<(B * H * 4096 + 255) / 256, 256, 0, stream>>>(dots_f, B * H * 4096);
    gemm_bf16_wmma<1, false, true><<<dim3(1, B * H, 8), 128, 0, stream>>>(
        vT_bf, kT_bf, nullptr, nullptr, dots_f, nullptr,
        64, 64, 512 /*K chunk*/, 4096, 4096, 64, 262144ull, 262144ull, 1, 4096ull, 0ull, 1.f);
    castscale_kernel<<<(B * H * 4096 + 255) / 256, 256, 0, stream>>>(
        dots_f, dots_bf, 1.f / 4096.f, B * H * 4096);
    // attn[b,n,h*64+e] = sum_d q[n,d]*dotsT[e,d] : A=q, W=dotsT
    gemm_bf16_wmma<2, false, false><<<dim3((4096 / 32) / 8, B * H), 256, 0, stream>>>(
        q_bf, dots_bf, nullptr, nullptr, nullptr, attn_bf,
        4096, 64, 64, 64, 64, 512, 262144ull, 4096ull, 8, 2097152ull, 64ull, 1.f);
    // out-proj + bias + residual -> xbuf (f32)
    gemm_bf16_wmma<2, false, false><<<dim3((R / 32) * (256 / 64) / 8, 1), 256, 0, stream>>>(
        attn_bf, wo_bf + (ull)l * 131072, bout + l * 256, xbuf, xbuf, nullptr,
        R, 256, 512, 512, 512, 256, 0ull, 0ull, 1, 0ull, 0ull, 1.f);

    // ---- FFN block ----
    ln_cast_kernel<<<R * 32 / 256, 256, 0, stream>>>(xbuf, ln2g + l * 256, ln2b + l * 256,
                                                     ln_bf, R);
    // fused FFN1 + GEGLU (RT=1): a- and g- column tiles, writes gelu(a)*g bf16 [R,1024]
    gemm_bf16_wmma<1, true, false><<<dim3((R / 16) * (1024 / 64) / 8, 1), 256, 0, stream>>>(
        ln_bf, w1_bf + (ull)l * 524288, b1 + l * 2048, nullptr, nullptr, gg_bf,
        R, 1024, 256, 256, 256, 1024, 0ull, 0ull, 1, 0ull, 0ull, 1.f);
    // FFN2 + bias + residual -> xbuf (f32)
    gemm_bf16_wmma<2, false, false><<<dim3((R / 32) * (256 / 64) / 8, 1), 256, 0, stream>>>(
        gg_bf, w2_bf + (ull)l * 262144, b2 + l * 256, xbuf, xbuf, nullptr,
        R, 256, 1024, 1024, 1024, 256, 0ull, 0ull, 1, 0ull, 0ull, 1.f);
  }
}
